// ImageEncoder_64587718197644
// MI455X (gfx1250) — matrix operand order repaired and checked by reading, not compiled, not run on hardware
//
#include <hip/hip_runtime.h>
#include <stdint.h>

// ---------------------------------------------------------------------------
// ViT-Base image encoder for gfx1250 (MI455X).
// bf16 WMMA for all matmuls, Tensor Data Mover (TDM) for all tile staging.
// ---------------------------------------------------------------------------

#define BATCH 32
#define SEQ   197          // tokens per image
#define WID   768
#define NLAY  12
#define NHEAD 12
#define HDIM  64
#define FFD   3072
#define EMB   512
#define NPATCH 196         // 14*14
#define MTOK  (BATCH * SEQ)      // 6304
#define MPATCH (BATCH * NPATCH)  // 6272

typedef __attribute__((ext_vector_type(16))) __bf16 v16bf;
typedef __attribute__((ext_vector_type(8)))  float  v8f;
typedef __attribute__((ext_vector_type(4)))  uint32_t u32x4;
typedef __attribute__((ext_vector_type(8)))  int      i32x8;
typedef __attribute__((ext_vector_type(4)))  int      i32x4;

union FragAB { v16bf v; uint32_t u[8]; };

__device__ __forceinline__ uint16_t bf16_of(float f) {
    uint32_t u = __float_as_uint(f);
    uint32_t r = u + 0x7FFFu + ((u >> 16) & 1u);   // round-to-nearest-even
    if ((u & 0x7F800000u) == 0x7F800000u) r = u;   // inf/nan passthrough
    return (uint16_t)(r >> 16);
}

// Low 32 bits of a flat shared-aperture address == LDS byte offset.
__device__ __forceinline__ uint32_t lds_off(const void* p) {
    return (uint32_t)(uintptr_t)p;
}

// ---------------------------------------------------------------------------
// TDM 2-D tile load, global -> LDS (ISA 08_async_tensor: D# groups 0..3).
// data_size = 2 bytes (bf16). Rows beyond tensor_d1 / cols beyond tensor_d0
// are zero-filled by hardware. pad_interval/pad_amount insert LDS row padding:
//   pad_interval: 0=2,1=4,2=8,3=16,4=32,5=64,... DWORDs between pads
//   pad_amount:   0=1,1=2,...  DWORDs of padding inserted
// Issue once per workgroup (wave 0); completion via s_wait_tensorcnt.
// ---------------------------------------------------------------------------
__device__ __forceinline__ void tdm_load_2d(
    uint32_t lds_byte_off, const void* gaddr,
    uint32_t tensor_d0, uint32_t tensor_d1,   // remaining extents from tile start (elems)
    uint32_t tile_d0,  uint32_t tile_d1,      // tile shape (elems)
    uint32_t stride0,                          // global row stride (elems)
    uint32_t pad_interval, uint32_t pad_amount)
{
    uint64_t ga = (uint64_t)(uintptr_t)gaddr;
    u32x4 g0;
    g0[0] = 1u;                                   // count=1 (valid user descriptor)
    g0[1] = lds_byte_off;                         // lds_addr
    g0[2] = (uint32_t)ga;                         // global_addr[31:0]
    g0[3] = (uint32_t)((ga >> 32) & 0x1FFFFFFu)   // global_addr[56:32]
          | (2u << 30);                           // type=2 ("image")
    i32x8 g1;
    g1[0] = (int)((1u << 16)                      // data_size = 1 -> 2 bytes
                | (1u << 20)                      // pad_enable
                | (pad_interval << 22)
                | (pad_amount << 25));
    g1[1] = (int)((tensor_d0 & 0xFFFFu) << 16);                       // dim0[15:0]
    g1[2] = (int)((tensor_d0 >> 16) | ((tensor_d1 & 0xFFFFu) << 16)); // dim0[31:16] | dim1[15:0]
    g1[3] = (int)((tensor_d1 >> 16) | (tile_d0 << 16));               // dim1[31:16] | tile0
    g1[4] = (int)(tile_d1 & 0xFFFFu);             // tile1 (tile2 = 0 -> 2-D)
    g1[5] = (int)stride0;                         // tensor_dim0_stride[31:0]
    g1[6] = 0;                                    // stride[47:32] | dim1_stride lo
    g1[7] = 0;
    i32x4 gz = {0, 0, 0, 0};                      // groups 2/3 unused (2-D tensor)
    asm volatile("tensor_load_to_lds %0, %1, %2, %3"
                 :: "s"(g0), "s"(g1), "s"(gz), "s"(gz) : "memory");
}

// ---------------------------------------------------------------------------
// Generic bf16 GEMM:  out[M,N] (+bias)(+gelu)(+residual) = A[M,K] @ B[K,N]
// Block tile 128x128, 8 waves (2M x 4N), wave tile 64x32 = 4x2 WMMAs of
// v_wmma_f32_16x16x32_bf16. K multiple of 32, N multiple of 128.
// Double-buffered TDM staging: DMA of tile i+1 overlaps compute of tile i.
// ---------------------------------------------------------------------------
#define LDA_P 40    // LDS A row stride (bf16): 64B row + 16B pad  (pad_int 3, amt 3)
#define LDB_P 136   // LDS B row stride (bf16): 256B row + 16B pad (pad_int 5, amt 3)

template<bool BIAS, bool GELU, bool RESID, bool ROWMAP, bool OUTF, bool OUTB>
__global__ __launch_bounds__(256)
void gemm_bf16(const uint16_t* __restrict__ A,   // [M][K] bf16
               const uint16_t* __restrict__ Bm,  // [K][N] bf16
               const float*    __restrict__ bias,// [N]
               float*          __restrict__ outF,
               uint16_t*       __restrict__ outB,
               int M, int N, int K)
{
    __shared__ uint16_t sA[2][128 * LDA_P];
    __shared__ uint16_t sB[2][32  * LDB_P];

    const int tid  = threadIdx.x;
    const int lane = tid & 31;
    const int wave = tid >> 5;
    const int hl   = lane >> 4;      // wave half (0/1)
    const int lm   = lane & 15;
    const int m0b  = blockIdx.y * 128;
    const int n0b  = blockIdx.x * 128;
    const int wm   = (wave & 1) * 64;
    const int wn   = (wave >> 1) * 32;

    v8f acc[4][2];
    #pragma unroll
    for (int i = 0; i < 4; ++i)
        #pragma unroll
        for (int j = 0; j < 2; ++j)
            #pragma unroll
            for (int e = 0; e < 8; ++e) acc[i][j][e] = 0.0f;

    const int nK = K / 32;
    if (wave == 0) {
        // prime tile 0
        tdm_load_2d(lds_off(&sA[0][0]), A + (size_t)m0b * K,
                    (uint32_t)K, (uint32_t)(M - m0b), 32, 128, (uint32_t)K, 3, 3);
        tdm_load_2d(lds_off(&sB[0][0]), Bm + n0b,
                    (uint32_t)(N - n0b), (uint32_t)K, 128, 32, (uint32_t)N, 5, 3);
    }

    for (int it = 0; it < nK; ++it) {
        const int buf = it & 1;
        const int k0  = it * 32;
        if (wave == 0) {
            if (it + 1 < nK) {
                const int k1 = k0 + 32;
                tdm_load_2d(lds_off(&sA[buf ^ 1][0]), A + (size_t)m0b * K + k1,
                            (uint32_t)(K - k1), (uint32_t)(M - m0b), 32, 128,
                            (uint32_t)K, 3, 3);
                tdm_load_2d(lds_off(&sB[buf ^ 1][0]), Bm + (size_t)k1 * N + n0b,
                            (uint32_t)(N - n0b), (uint32_t)(K - k1), 128, 32,
                            (uint32_t)N, 5, 3);
                __builtin_amdgcn_s_wait_tensorcnt(2);   // tile `it` done, i+1 in flight
            } else {
                __builtin_amdgcn_s_wait_tensorcnt(0);
            }
        }
        __syncthreads();                                // tile `it` visible to all waves

        // ---- fragments per ISA 7.12.2 layouts ----
        FragAB fa[4], fb[2];
        #pragma unroll
        for (int sm = 0; sm < 4; ++sm) {
            int m = wm + sm * 16 + lm;
            #pragma unroll
            for (int vv = 0; vv < 8; ++vv) {
                int kb = (vv >> 2) * 16 + hl * 8 + (vv & 3) * 2;
                fa[sm].u[vv] = *(const uint32_t*)(&sA[buf][0] + m * LDA_P + kb);
            }
        }
        #pragma unroll
        for (int sn = 0; sn < 2; ++sn) {
            int nb = wn + sn * 16;
            #pragma unroll
            for (int vv = 0; vv < 8; ++vv) {
                int kb = (vv >> 2) * 16 + hl * 8 + (vv & 3) * 2;
                const uint16_t* p = &sB[buf][0] + kb * LDB_P + nb + lm;
                fb[sn].u[vv] = (uint32_t)p[0] | ((uint32_t)p[LDB_P] << 16);
            }
        }
        #pragma unroll
        for (int sm = 0; sm < 4; ++sm)
            #pragma unroll
            for (int sn = 0; sn < 2; ++sn)
                acc[sm][sn] = __builtin_amdgcn_wmma_f32_16x16x32_bf16(
                    false, fa[sm].v, false, fb[sn].v,
                    (short)0, acc[sm][sn], false, false);
        __syncthreads();                                // reads done before buf reused
    }

    // ---- epilogue ----
    #pragma unroll
    for (int sm = 0; sm < 4; ++sm) {
        #pragma unroll
        for (int sn = 0; sn < 2; ++sn) {
            #pragma unroll
            for (int vv = 0; vv < 8; ++vv) {
                int row = m0b + wm + sm * 16 + vv + 8 * hl;
                int col = n0b + wn + sn * 16 + lm;
                if (row < M) {
                    float c = acc[sm][sn][vv];
                    if (BIAS) c += bias[col];
                    if (GELU) c = 0.5f * c * (1.0f + erff(c * 0.70710678f));
                    int orow = ROWMAP ? (row + row / NPATCH + 1) : row;
                    size_t idx = (size_t)orow * N + col;
                    if (RESID) c += outF[idx];
                    if (OUTF)  outF[idx] = c;
                    if (OUTB)  outB[idx] = bf16_of(c);
                }
            }
        }
    }
}

// ---------------------------------------------------------------------------
// Fused attention per (b, h): scores = Q Kt /8 -> softmax -> P V, all WMMA.
// Q/K/V tiles staged by TDM with hardware zero-fill beyond row 197.
// Dynamic LDS layout (273.0 KB total):
//   [0, 186368)        : scores f32 [208][224]  (V bf16 [224][68] overlays in phase 3)
//   [186368, 279552)   : probs bf16 [208][224]  (Q,K bf16 [208][68] overlay in phase 1)
// ---------------------------------------------------------------------------
#define SC_W 224
#define SC_H 208
#define Q_P  68     // 128B row + 8B pad (pad_interval 4 = 32 dwords, pad_amount 1 = 2 dwords)

__global__ __launch_bounds__(256)
void attn_kernel(const uint16_t* __restrict__ qg,
                 const uint16_t* __restrict__ kg,
                 const uint16_t* __restrict__ vg,
                 uint16_t*       __restrict__ ao)
{
    extern __shared__ char smem[];
    float*    sc = (float*)smem;                           // [208][224] scores
    uint16_t* pr = (uint16_t*)(smem + SC_H * SC_W * 4);    // [208][224] probs
    uint16_t* qs = pr;                                     // Q [208][Q_P]
    uint16_t* ks = pr + SC_H * Q_P;                        // K [208][Q_P]
    uint16_t* vs = (uint16_t*)smem;                        // V [224][Q_P] (phase 3)

    const int bh = blockIdx.x;
    const int b  = bh / NHEAD, hh = bh % NHEAD;
    const size_t base = ((size_t)b * SEQ) * WID + hh * HDIM;
    const int tid = threadIdx.x, lane = tid & 31, wave = tid >> 5;
    const int hl = lane >> 4, lm = lane & 15;

    // phase 0: TDM-stage Q and K (rows >= 197 zero-filled by tensor_dim1)
    if (wave == 0) {
        tdm_load_2d(lds_off(qs), qg + base, (uint32_t)(WID - hh * HDIM), SEQ,
                    HDIM, SC_H, WID, 4, 1);
        tdm_load_2d(lds_off(ks), kg + base, (uint32_t)(WID - hh * HDIM), SEQ,
                    HDIM, SC_H, WID, 4, 1);
        __builtin_amdgcn_s_wait_tensorcnt(0);
    }
    __syncthreads();

    // phase 1: scores = Q @ K^T * 1/8
    for (int tile = wave; tile < 13 * 13; tile += 8) {
        int s0 = (tile / 13) * 16, t0 = (tile % 13) * 16;
        v8f acc;
        #pragma unroll
        for (int e = 0; e < 8; ++e) acc[e] = 0.0f;
        #pragma unroll
        for (int d0 = 0; d0 < HDIM; d0 += 32) {
            FragAB fa, fb;
            int m = s0 + lm;
            #pragma unroll
            for (int vv = 0; vv < 8; ++vv) {
                int kb = (vv >> 2) * 16 + hl * 8 + (vv & 3) * 2;
                fa.u[vv] = *(const uint32_t*)(qs + m * Q_P + d0 + kb);
            }
            int t = t0 + lm;
            #pragma unroll
            for (int vv = 0; vv < 8; ++vv) {
                int kb = (vv >> 2) * 16 + hl * 8 + (vv & 3) * 2;
                fb.u[vv] = *(const uint32_t*)(ks + t * Q_P + d0 + kb);
            }
            acc = __builtin_amdgcn_wmma_f32_16x16x32_bf16(
                false, fa.v, false, fb.v, (short)0, acc, false, false);
        }
        #pragma unroll
        for (int vv = 0; vv < 8; ++vv)
            sc[(s0 + vv + 8 * hl) * SC_W + t0 + lm] = acc[vv] * 0.125f;
    }
    __syncthreads();

    // phase 2: softmax rows, re-quantize to bf16 probs (zero the padding)
    if (tid < SC_H) {
        int s = tid;
        if (s < SEQ) {
            float mx = -1e30f;
            for (int t = 0; t < SEQ; ++t) mx = fmaxf(mx, sc[s * SC_W + t]);
            float sum = 0.0f;
            for (int t = 0; t < SEQ; ++t) {
                float e = __expf(sc[s * SC_W + t] - mx);
                sc[s * SC_W + t] = e;
                sum += e;
            }
            float inv = 1.0f / sum;
            for (int t = 0; t < SEQ; ++t)  pr[s * SC_W + t] = bf16_of(sc[s * SC_W + t] * inv);
            for (int t = SEQ; t < SC_W; ++t) pr[s * SC_W + t] = 0;
        } else {
            for (int t = 0; t < SC_W; ++t) pr[s * SC_W + t] = 0;
        }
    }
    __syncthreads();                                   // sc now dead -> reuse for V

    // phase 2b: TDM-stage V over the scores region; tile_dim1=224 so the padded
    // K-extent rows 197..223 are hardware zero-filled (no stale-bit NaNs).
    if (wave == 0) {
        tdm_load_2d(lds_off(vs), vg + base, (uint32_t)(WID - hh * HDIM), SEQ,
                    HDIM, SC_W, WID, 4, 1);
        __builtin_amdgcn_s_wait_tensorcnt(0);
    }
    __syncthreads();

    // phase 3: out = P @ V
    for (int tile = wave; tile < 13 * 4; tile += 8) {
        int s0 = (tile >> 2) * 16, d0 = (tile & 3) * 16;
        v8f acc;
        #pragma unroll
        for (int e = 0; e < 8; ++e) acc[e] = 0.0f;
        for (int k0 = 0; k0 < SC_W; k0 += 32) {
            FragAB fa, fb;
            int m = s0 + lm;
            #pragma unroll
            for (int vv = 0; vv < 8; ++vv) {
                int kb = (vv >> 2) * 16 + hl * 8 + (vv & 3) * 2;
                fa.u[vv] = *(const uint32_t*)(pr + m * SC_W + k0 + kb);
            }
            #pragma unroll
            for (int vv = 0; vv < 8; ++vv) {
                int kb = (vv >> 2) * 16 + hl * 8 + (vv & 3) * 2;
                const uint16_t* p = vs + (k0 + kb) * Q_P + d0 + lm;
                fb.u[vv] = (uint32_t)p[0] | ((uint32_t)p[Q_P] << 16);
            }
            acc = __builtin_amdgcn_wmma_f32_16x16x32_bf16(
                false, fa.v, false, fb.v, (short)0, acc, false, false);
        }
        #pragma unroll
        for (int vv = 0; vv < 8; ++vv) {
            int s = s0 + vv + 8 * hl;
            if (s < SEQ)
                ao[base + (size_t)s * WID + d0 + lm] = bf16_of(acc[vv]);
        }
    }
}

// ---------------------------------------------------------------------------
// LayerNorm over last dim (768), bf16 output. One block of 256 per row.
// ---------------------------------------------------------------------------
__global__ __launch_bounds__(256)
void layernorm_kernel(const float* __restrict__ x,
                      const float* __restrict__ g,
                      const float* __restrict__ bta,
                      uint16_t*    __restrict__ y)
{
    __shared__ float red[256];
    const int row = blockIdx.x, tid = threadIdx.x;
    const float* xr = x + (size_t)row * WID;
    float v0 = xr[tid], v1 = xr[tid + 256], v2 = xr[tid + 512];
    red[tid] = v0 + v1 + v2;
    __syncthreads();
    for (int s = 128; s > 0; s >>= 1) { if (tid < s) red[tid] += red[tid + s]; __syncthreads(); }
    float mean = red[0] * (1.0f / WID);
    __syncthreads();
    float d0 = v0 - mean, d1 = v1 - mean, d2 = v2 - mean;
    red[tid] = d0 * d0 + d1 * d1 + d2 * d2;
    __syncthreads();
    for (int s = 128; s > 0; s >>= 1) { if (tid < s) red[tid] += red[tid + s]; __syncthreads(); }
    float rs = rsqrtf(red[0] * (1.0f / WID) + 1e-5f);
    uint16_t* yr = y + (size_t)row * WID;
    yr[tid]       = bf16_of(d0 * rs * g[tid]       + bta[tid]);
    yr[tid + 256] = bf16_of(d1 * rs * g[tid + 256] + bta[tid + 256]);
    yr[tid + 512] = bf16_of(d2 * rs * g[tid + 512] + bta[tid + 512]);
}

// ---------------------------------------------------------------------------
// Small elementwise kernels
// ---------------------------------------------------------------------------
__global__ void im2col_kernel(const float* __restrict__ x, uint16_t* __restrict__ Ap)
{
    int idx = blockIdx.x * 256 + threadIdx.x;
    if (idx >= MPATCH * WID) return;
    int row = idx / WID, ck = idx % WID;
    int bb = row / NPATCH, p = row % NPATCH;
    int py = p / 14, px = p % 14;
    int c = ck >> 8, rr = ck & 255, iy = rr >> 4, ix = rr & 15;
    Ap[idx] = bf16_of(x[(((size_t)bb * 3 + c) * 224 + py * 16 + iy) * 224 + px * 16 + ix]);
}

__global__ void cvt_kernel(const float* __restrict__ s, uint16_t* __restrict__ d, int n)
{
    int i = blockIdx.x * 256 + threadIdx.x;
    if (i < n) d[i] = bf16_of(s[i]);
}

// conv_w [O=768][K=768] row-major -> [K][O] bf16
__global__ void cvt_t_kernel(const float* __restrict__ s, uint16_t* __restrict__ d)
{
    int i = blockIdx.x * 256 + threadIdx.x;
    if (i >= WID * WID) return;
    int kk = i / WID, o = i % WID;
    d[i] = bf16_of(s[(size_t)o * WID + kk]);
}

__global__ void add_pe_kernel(float* __restrict__ h,
                              const float* __restrict__ cls,
                              const float* __restrict__ pe)
{
    int row = blockIdx.x;
    int s = row % SEQ;
    for (int c = threadIdx.x; c < WID; c += 256) {
        size_t idx = (size_t)row * WID + c;
        if (s == 0) h[idx] = cls[c] + pe[c];
        else        h[idx] += pe[(size_t)s * WID + c];
    }
}

// head: out[b] = normalize(h[b,0,:] @ proj)
__global__ __launch_bounds__(512)
void head_kernel(const float* __restrict__ h,
                 const float* __restrict__ proj,
                 float* __restrict__ out)
{
    __shared__ float red[512];
    const int b = blockIdx.x, e = threadIdx.x;
    const float* hr = h + (size_t)b * SEQ * WID;   // row 0 (cls token)
    float acc = 0.0f;
    for (int k2 = 0; k2 < WID; ++k2)
        acc += hr[k2] * proj[(size_t)k2 * EMB + e];
    red[e] = acc * acc;
    __syncthreads();
    for (int s = 256; s > 0; s >>= 1) { if (e < s) red[e] += red[e + s]; __syncthreads(); }
    out[(size_t)b * EMB + e] = acc * rsqrtf(red[0]);
}

// ---------------------------------------------------------------------------
// Host orchestration
// ---------------------------------------------------------------------------
extern "C" void kernel_launch(void* const* d_in, const int* in_sizes, int n_in,
                              void* d_out, int out_size, void* d_ws, size_t ws_size,
                              hipStream_t stream)
{
    (void)in_sizes; (void)n_in; (void)out_size; (void)ws_size;

    const float* x      = (const float*)d_in[0];
    const float* conv_w = (const float*)d_in[1];
    const float* conv_b = (const float*)d_in[2];
    const float* clsX   = (const float*)d_in[3];
    const float* pe     = (const float*)d_in[4];
    const float* ln1_g  = (const float*)d_in[5];
    const float* ln1_b  = (const float*)d_in[6];
    const float* wq     = (const float*)d_in[7];
    const float* bq     = (const float*)d_in[8];
    const float* wk     = (const float*)d_in[9];
    const float* bk     = (const float*)d_in[10];
    const float* wv     = (const float*)d_in[11];
    const float* bv     = (const float*)d_in[12];
    const float* wo     = (const float*)d_in[13];
    const float* bo     = (const float*)d_in[14];
    const float* ln2_g  = (const float*)d_in[15];
    const float* ln2_b  = (const float*)d_in[16];
    const float* w1     = (const float*)d_in[17];
    const float* b1     = (const float*)d_in[18];
    const float* w2     = (const float*)d_in[19];
    const float* b2     = (const float*)d_in[20];
    const float* proj   = (const float*)d_in[21];
    float* out = (float*)d_out;

    // ---- workspace layout (~111 MB) ----
    char* ws = (char*)d_ws;
    size_t off = 0;
    float*    h   = (float*)(ws + off);    off += (size_t)MTOK * WID * 4;   // residual f32
    uint16_t* yb  = (uint16_t*)(ws + off); off += (size_t)MTOK * WID * 2;   // LN out bf16
    uint16_t* qb  = (uint16_t*)(ws + off); off += (size_t)MTOK * WID * 2;
    uint16_t* kb  = (uint16_t*)(ws + off); off += (size_t)MTOK * WID * 2;
    uint16_t* vb  = (uint16_t*)(ws + off); off += (size_t)MTOK * WID * 2;
    uint16_t* aob = (uint16_t*)(ws + off); off += (size_t)MTOK * WID * 2;
    uint16_t* ffb = (uint16_t*)(ws + off); off += (size_t)MTOK * FFD * 2;   // FF intermediate
    uint16_t* wbuf = (uint16_t*)(ws + off); off += (size_t)FFD * WID * 2;   // bf16 weight slab
    uint16_t* apb = ffb;   // im2col buffer overlays FF buffer (disjoint lifetimes)

    const dim3 blk(256);
    const dim3 gW(WID / 128, (MTOK + 127) / 128);     // N=768, M=6304
    const dim3 gF(FFD / 128, (MTOK + 127) / 128);     // N=3072
    const dim3 gP(WID / 128, MPATCH / 128);           // patch GEMM, M=6272

    // 1) patch embedding = im2col GEMM (ROWMAP interleaves past CLS slot)
    im2col_kernel<<<(MPATCH * WID + 255) / 256, blk, 0, stream>>>(x, apb);
    cvt_t_kernel<<<(WID * WID + 255) / 256, blk, 0, stream>>>(conv_w, wbuf);
    gemm_bf16<true, false, false, true, true, false><<<gP, blk, 0, stream>>>(
        apb, wbuf, conv_b, h, nullptr, MPATCH, WID, WID);

    // 2) cls token + positional encoding
    add_pe_kernel<<<MTOK, blk, 0, stream>>>(h, clsX, pe);

    const int WW = WID * WID;
    const size_t attn_lds = (size_t)SC_H * SC_W * 4 + (size_t)SC_H * SC_W * 2; // 273 KB

    // 3) transformer blocks
    for (int l = 0; l < NLAY; ++l) {
        layernorm_kernel<<<MTOK, blk, 0, stream>>>(h, ln1_g + l * WID, ln1_b + l * WID, yb);

        cvt_kernel<<<(WW + 255) / 256, blk, 0, stream>>>(wq + (size_t)l * WW, wbuf, WW);
        gemm_bf16<true, false, false, false, false, true><<<gW, blk, 0, stream>>>(
            yb, wbuf, bq + l * WID, nullptr, qb, MTOK, WID, WID);
        cvt_kernel<<<(WW + 255) / 256, blk, 0, stream>>>(wk + (size_t)l * WW, wbuf, WW);
        gemm_bf16<true, false, false, false, false, true><<<gW, blk, 0, stream>>>(
            yb, wbuf, bk + l * WID, nullptr, kb, MTOK, WID, WID);
        cvt_kernel<<<(WW + 255) / 256, blk, 0, stream>>>(wv + (size_t)l * WW, wbuf, WW);
        gemm_bf16<true, false, false, false, false, true><<<gW, blk, 0, stream>>>(
            yb, wbuf, bv + l * WID, nullptr, vb, MTOK, WID, WID);

        attn_kernel<<<BATCH * NHEAD, blk, attn_lds, stream>>>(qb, kb, vb, aob);

        cvt_kernel<<<(WW + 255) / 256, blk, 0, stream>>>(wo + (size_t)l * WW, wbuf, WW);
        gemm_bf16<true, false, true, false, true, false><<<gW, blk, 0, stream>>>(
            aob, wbuf, bo + l * WID, h, nullptr, MTOK, WID, WID);

        layernorm_kernel<<<MTOK, blk, 0, stream>>>(h, ln2_g + l * WID, ln2_b + l * WID, yb);

        cvt_kernel<<<(WID * FFD + 255) / 256, blk, 0, stream>>>(w1 + (size_t)l * WID * FFD, wbuf, WID * FFD);
        gemm_bf16<true, true, false, false, false, true><<<gF, blk, 0, stream>>>(
            yb, wbuf, b1 + l * FFD, nullptr, ffb, MTOK, FFD, WID);

        cvt_kernel<<<(WID * FFD + 255) / 256, blk, 0, stream>>>(w2 + (size_t)l * WID * FFD, wbuf, WID * FFD);
        gemm_bf16<true, false, true, false, true, false><<<gW, blk, 0, stream>>>(
            ffb, wbuf, b2 + l * WID, h, nullptr, MTOK, WID, FFD);
    }

    // 4) projection head + L2 normalize
    head_kernel<<<BATCH, dim3(512), 0, stream>>>(h, proj, out);
}
